// SparseCrossAttention_38156489458187
// MI455X (gfx1250) — matrix operand order repaired and checked by reading, not compiled, not run on hardware
//
#include <hip/hip_runtime.h>
#include <hip/hip_bf16.h>

#define DEVI __device__ __forceinline__

typedef __attribute__((ext_vector_type(16))) __bf16 bf16x16;
typedef __attribute__((ext_vector_type(8)))  float  f32x8;

namespace cfg {
constexpr int B = 8, S = 2048, DIM = 2048;
constexpr int DCTX = 1024;
constexpr int H = 8, DH = 64, INNER = H * DH;   // 512
constexpr int KV = 256;                         // M*N
constexpr int TOPK = 32;
constexpr float EPS = 1e-5f;
}

// ---------------------------------------------------------------------------
// WMMA fragment helpers (CDNA5 ISA 7.12.2 layouts, wave32)
// A 16x32 bf16: lane L<16 -> row L, K pairs {0..7} in v0-3 and {16..23} in v4-7;
//               lane L>=16 -> row L-16, K {8..15} and {24..31}.
// B 32x16 bf16: lanes 0-15 hold K {0..7,16..23} of column (lane), lanes 16-31 K {8..15,24..31}.
// C/D f32 16x16: vgpr r: lanes 0-15 -> M=r, lanes 16-31 -> M=r+8; N = lane&15.
// ---------------------------------------------------------------------------
DEVI bf16x16 ld_a_frag(const __bf16* base, int lda, int k0) {
  const int lane = threadIdx.x & 31;
  const __bf16* p = base + (size_t)(lane & 15) * lda + k0 + ((lane >> 4) << 3);
  bf16x16 f;
  ((uint4*)&f)[0] = *(const uint4*)p;         // K = kb .. kb+7
  ((uint4*)&f)[1] = *(const uint4*)(p + 16);  // K = kb+16 .. kb+23
  return f;
}
// Bt is stored [N, K] (i.e. B transposed) so each lane reads K in two contiguous runs of 8.
DEVI bf16x16 ld_b_frag(const __bf16* base, int ldk, int k0) {
  const int lane = threadIdx.x & 31;
  const __bf16* p = base + (size_t)(lane & 15) * ldk + k0 + ((lane >> 4) << 3);
  bf16x16 f;
  ((uint4*)&f)[0] = *(const uint4*)p;
  ((uint4*)&f)[1] = *(const uint4*)(p + 16);
  return f;
}
DEVI f32x8 wmma_bf16(bf16x16 a, bf16x16 b, f32x8 c) {
  return __builtin_amdgcn_wmma_f32_16x16x32_bf16(false, a, false, b, (short)0, c,
                                                 false, false);
}

// ---------------------------------------------------------------------------
// LayerNorm (fp32 in -> bf16 out), one 256-thread block (8 waves) per row.
// ---------------------------------------------------------------------------
template <int DIMV>
__global__ __launch_bounds__(256) void ln_to_bf16(const float* __restrict__ X,
                                                  const float* __restrict__ G,
                                                  const float* __restrict__ Bi,
                                                  __bf16* __restrict__ Y) {
  const size_t row = blockIdx.x;
  const float* x = X + row * DIMV;
  __bf16* y = Y + row * DIMV;
  __shared__ float rs[8], rss[8];
  float s = 0.f, ss = 0.f;
  for (int i = threadIdx.x; i < DIMV; i += 256) {
    const float v = x[i];
    s += v;
    ss += v * v;
  }
#pragma unroll
  for (int o = 16; o > 0; o >>= 1) {
    s += __shfl_xor(s, o, 32);
    ss += __shfl_xor(ss, o, 32);
  }
  const int wid = threadIdx.x >> 5;
  if ((threadIdx.x & 31) == 0) { rs[wid] = s; rss[wid] = ss; }
  __syncthreads();
  if (threadIdx.x == 0) {
    float a = 0.f, c = 0.f;
#pragma unroll
    for (int i = 0; i < 8; ++i) { a += rs[i]; c += rss[i]; }
    rs[0] = a;
    rss[0] = c;
  }
  __syncthreads();
  const float mu = rs[0] / DIMV;
  const float var = rss[0] / DIMV - mu * mu;
  const float rstd = rsqrtf(var + cfg::EPS);
  for (int i = threadIdx.x; i < DIMV; i += 256)
    y[i] = (__bf16)((x[i] - mu) * rstd * G[i] + Bi[i]);
}

// ---------------------------------------------------------------------------
// Convert fp32 weight [K,N] -> bf16 transposed [N,K] (one-time, 6MB total).
// ---------------------------------------------------------------------------
__global__ __launch_bounds__(256) void w_transpose_bf16(
    const float* __restrict__ W, __bf16* __restrict__ Wt, int K, int Nn) {
  const size_t idx = (size_t)blockIdx.x * blockDim.x + threadIdx.x;
  if (idx >= (size_t)K * Nn) return;
  const int k = (int)(idx % K);
  const int n = (int)(idx / K);
  Wt[idx] = (__bf16)W[(size_t)k * Nn + n];
}

// ---------------------------------------------------------------------------
// GEMM: C[Mr,Nc] = A[Mr,K](bf16 row-major) * Bt[Nc,K](bf16, B transposed).
// One wave owns a (MT*16)x64 tile; MT=4 gives 16 WMMAs per 32-wide k-step
// from 8KB of fragment loads (~32 FLOP/byte from L2) and deep WMMA ILP.
// ---------------------------------------------------------------------------
template <typename OutT, int MT>
__global__ __launch_bounds__(256) void gemm_bf16_nt(
    const __bf16* __restrict__ A, const __bf16* __restrict__ Bt,
    OutT* __restrict__ C, int Mr, int Nc, int K) {
  const int wave = (int)(((size_t)blockIdx.x * blockDim.x + threadIdx.x) >> 5);
  const int tilesN = Nc >> 6;
  const int tm = wave / tilesN;
  const int tn = wave % tilesN;
  if (tm * (MT * 16) >= Mr) return;
  f32x8 acc[MT][4] = {};
  for (int k0 = 0; k0 < K; k0 += 32) {
    bf16x16 a[MT];
#pragma unroll
    for (int m = 0; m < MT; ++m)
      a[m] = ld_a_frag(A + (size_t)(tm * MT * 16 + m * 16) * K, K, k0);
    if (k0 + 64 < K)  // emits global_prefetch_b8
      __builtin_prefetch(
          A + (size_t)(tm * MT * 16 + (threadIdx.x & 15)) * K + k0 + 64, 0, 1);
#pragma unroll
    for (int i = 0; i < 4; ++i) {
      const bf16x16 bfrag =
          ld_b_frag(Bt + (size_t)(tn * 64 + i * 16) * K, K, k0);
#pragma unroll
      for (int m = 0; m < MT; ++m) acc[m][i] = wmma_bf16(a[m], bfrag, acc[m][i]);
    }
  }
  const int lane = threadIdx.x & 31;
  const int colb = tn * 64 + (lane & 15);
  const int mbase = tm * MT * 16 + ((lane >> 4) << 3);
#pragma unroll
  for (int m = 0; m < MT; ++m)
#pragma unroll
    for (int i = 0; i < 4; ++i)
#pragma unroll
      for (int r = 0; r < 8; ++r)
        C[(size_t)(mbase + m * 16 + r) * Nc + colb + i * 16] =
            (OutT)acc[m][i][r];
}

// ---------------------------------------------------------------------------
// Sparse cross-attention core. Block = 128 threads (4 waves); block handles
// one (b, h, 64-query group); each wave one 16-query tile.
// LDS: K tile [256][64] bf16 (staged via GLOBAL_LOAD_ASYNC_TO_LDS_B128,
// ASYNCcnt-tracked), V^T tile [64][256] bf16, per-wave sim[16][256] f32,
// per-wave row stats (max, threshold, 1/Z).
// ---------------------------------------------------------------------------
__global__ __launch_bounds__(128) void sparse_attn(
    const __bf16* __restrict__ qbuf, const __bf16* __restrict__ kvbuf,
    __bf16* __restrict__ obuf) {
  using namespace cfg;
  extern __shared__ char smem[];
  __bf16* kt = (__bf16*)smem;                // [KV][DH]
  __bf16* vt = kt + KV * DH;                 // [DH][KV] (transposed)
  float* simAll = (float*)(vt + DH * KV);    // [4][16][KV]
  float* stats = simAll + 4 * 16 * KV;       // [4][16][3]

  const int sTiles = S / 64;  // 32
  const int b = blockIdx.x / (H * sTiles);
  const int hz = blockIdx.x % (H * sTiles);
  const int h = hz / sTiles;
  const int sg = hz % sTiles;

  // Stage K directly memory->LDS with async-tensor path (no VGPR bounce);
  // stage V^T manually (transposed in flight).
  const __bf16* kvb = kvbuf + (size_t)b * KV * (2 * INNER);
  for (int t = threadIdx.x; t < KV * (DH / 8); t += blockDim.x) {
    const int j = t >> 3;
    const int dc = (t & 7) << 3;
    const __bf16* gk = &kvb[(size_t)j * (2 * INNER) + h * DH + dc];
    const unsigned ldsoff = (unsigned)(size_t)&kt[j * DH + dc];
    asm volatile("global_load_async_to_lds_b128 %0, %1, off"
                 :
                 : "v"(ldsoff), "v"(gk)
                 : "memory");
    const __bf16* vsrc = &kvb[(size_t)j * (2 * INNER) + INNER + h * DH + dc];
#pragma unroll
    for (int e = 0; e < 8; ++e) vt[(dc + e) * KV + j] = vsrc[e];
  }
  asm volatile("s_wait_asynccnt 0x0" ::: "memory");
  __syncthreads();

  const int wave = threadIdx.x >> 5;
  const int lane = threadIdx.x & 31;
  const int s0 = (sg * 4 + wave) * 16;
  float* sim = simAll + wave * 16 * KV;
  float* st = stats + wave * 16 * 3;

  // sim[16,256] = q_tile(16x64) @ K^T, via 16 N-tiles x 2 K-steps of WMMA.
  const __bf16* qtile = qbuf + (size_t)(b * S + s0) * INNER + h * DH;
  const bf16x16 a0 = ld_a_frag(qtile, INNER, 0);
  const bf16x16 a1 = ld_a_frag(qtile, INNER, 32);
  constexpr float scale = 0.125f;  // DH^-0.5
  for (int nt = 0; nt < KV / 16; ++nt) {
    f32x8 acc = {};
    acc = wmma_bf16(a0, ld_b_frag(kt + nt * 16 * DH, DH, 0), acc);
    acc = wmma_bf16(a1, ld_b_frag(kt + nt * 16 * DH, DH, 32), acc);
    const int col = nt * 16 + (lane & 15);
    const int mb = (lane >> 4) << 3;
#pragma unroll
    for (int r = 0; r < 8; ++r) sim[(mb + r) * KV + col] = acc[r] * scale;
  }
  __syncthreads();

  // Per-row top-32: lanes 0..15 each own one row. softmax(top_k scattered)
  // == exp(s-m)/Z masked by s >= t (t = 32nd largest).
  if (lane < 16) {
    const float* srow = sim + lane * KV;
    float heap[TOPK];
#pragma unroll
    for (int i = 0; i < TOPK; ++i) heap[i] = srow[i];
    float hmin = heap[0];
    int hidx = 0;
    for (int i = 1; i < TOPK; ++i)
      if (heap[i] < hmin) { hmin = heap[i]; hidx = i; }
    for (int j = TOPK; j < KV; ++j) {
      const float v = srow[j];
      if (v > hmin) {
        heap[hidx] = v;
        hmin = heap[0];
        hidx = 0;
        for (int i = 1; i < TOPK; ++i)
          if (heap[i] < hmin) { hmin = heap[i]; hidx = i; }
      }
    }
    const float thr = hmin;
    float mmax = heap[0];
    for (int i = 1; i < TOPK; ++i) mmax = fmaxf(mmax, heap[i]);
    float Z = 0.f;
    for (int j = 0; j < KV; ++j) {
      const float v = srow[j];
      if (v >= thr) Z += __expf(v - mmax);
    }
    st[lane * 3 + 0] = mmax;
    st[lane * 3 + 1] = thr;
    st[lane * 3 + 2] = 1.0f / Z;
  }
  __syncthreads();

  // out[16,64] = p[16,256] @ V[256,64]; p rebuilt as bf16 A-fragments on the fly.
  f32x8 oacc[4] = {};
  const int prow = lane & 15;
  const float pm = st[prow * 3 + 0];
  const float pt = st[prow * 3 + 1];
  const float pz = st[prow * 3 + 2];
  const float* srow = sim + prow * KV;
  for (int k0 = 0; k0 < KV; k0 += 32) {
    const int kb = k0 + ((lane >> 4) << 3);
    bf16x16 ap;
#pragma unroll
    for (int i = 0; i < 8; ++i) {
      const float v1 = srow[kb + i];
      const float v2 = srow[kb + 16 + i];
      ap[i] = (__bf16)(v1 >= pt ? __expf(v1 - pm) * pz : 0.0f);
      ap[i + 8] = (__bf16)(v2 >= pt ? __expf(v2 - pm) * pz : 0.0f);
    }
#pragma unroll
    for (int nt = 0; nt < 4; ++nt)
      oacc[nt] = wmma_bf16(ap, ld_b_frag(vt + nt * 16 * KV, KV, k0), oacc[nt]);
  }
  const int colb = h * DH + (lane & 15);
  const int mrow = b * S + s0 + ((lane >> 4) << 3);
#pragma unroll
  for (int nt = 0; nt < 4; ++nt)
#pragma unroll
    for (int r = 0; r < 8; ++r)
      obuf[(size_t)(mrow + r) * INNER + colb + nt * 16] = (__bf16)oacc[nt][r];
}

// ---------------------------------------------------------------------------
extern "C" void kernel_launch(void* const* d_in, const int* in_sizes, int n_in,
                              void* d_out, int out_size, void* d_ws,
                              size_t ws_size, hipStream_t stream) {
  using namespace cfg;
  const float* x = (const float*)d_in[0];
  const float* ctx = (const float*)d_in[1];
  const float* ln_g = (const float*)d_in[2];
  const float* ln_b = (const float*)d_in[3];
  const float* lnc_g = (const float*)d_in[4];
  const float* lnc_b = (const float*)d_in[5];
  const float* Wq = (const float*)d_in[6];
  const float* Wkv = (const float*)d_in[7];
  const float* Wout = (const float*)d_in[8];
  // d_in[9] = top_k; reference fixes it at 32 (cfg::TOPK).
  float* out = (float*)d_out;

  char* ws = (char*)d_ws;
  auto take = [&](size_t bytes) {
    char* p = ws;
    ws += (bytes + 255) & ~(size_t)255;
    return p;
  };
  __bf16* xn = (__bf16*)take((size_t)B * S * DIM * 2);          // 64 MB
  __bf16* cn = (__bf16*)take((size_t)B * KV * DCTX * 2);        // 4 MB
  __bf16* Wqt = (__bf16*)take((size_t)INNER * DIM * 2);         // 2 MB
  __bf16* Wkvt = (__bf16*)take((size_t)2 * INNER * DCTX * 2);   // 2 MB
  __bf16* Woutt = (__bf16*)take((size_t)DIM * INNER * 2);       // 2 MB
  __bf16* qb = (__bf16*)take((size_t)B * S * INNER * 2);        // 16 MB
  __bf16* kvb = (__bf16*)take((size_t)B * KV * 2 * INNER * 2);  // 4 MB
  __bf16* ob = (__bf16*)take((size_t)B * S * INNER * 2);        // 16 MB

  ln_to_bf16<DIM><<<B * S, 256, 0, stream>>>(x, ln_g, ln_b, xn);
  ln_to_bf16<DCTX><<<B * KV, 256, 0, stream>>>(ctx, lnc_g, lnc_b, cn);

  w_transpose_bf16<<<(DIM * INNER + 255) / 256, 256, 0, stream>>>(Wq, Wqt, DIM,
                                                                  INNER);
  w_transpose_bf16<<<(DCTX * 2 * INNER + 255) / 256, 256, 0, stream>>>(
      Wkv, Wkvt, DCTX, 2 * INNER);
  w_transpose_bf16<<<(INNER * DIM + 255) / 256, 256, 0, stream>>>(Wout, Woutt,
                                                                  INNER, DIM);

  {  // q = xn @ Wq : [B*S, INNER], 64x64 wave tiles
    const int waves = (B * S / 64) * (INNER / 64);  // 2048
    gemm_bf16_nt<__bf16, 4><<<waves * 32 / 256, 256, 0, stream>>>(
        xn, Wqt, qb, B * S, INNER, DIM);
  }
  {  // kv = cn @ Wkv : [B*KV, 2*INNER], small M -> 16x64 tiles for parallelism
    const int waves = (B * KV / 16) * (2 * INNER / 64);  // 2048
    gemm_bf16_nt<__bf16, 1><<<waves * 32 / 256, 256, 0, stream>>>(
        cn, Wkvt, kvb, B * KV, 2 * INNER, DCTX);
  }

  const size_t smem = (size_t)KV * DH * 2 + (size_t)DH * KV * 2 +
                      (size_t)4 * 16 * KV * 4 + (size_t)4 * 16 * 3 * 4;
  sparse_attn<<<B * H * (S / 64), 128, smem, stream>>>(qb, kvb, ob);

  {  // final: out = ob @ Wout : [B*S, DIM] fp32, 64x64 wave tiles
    const int waves = (B * S / 64) * (DIM / 64);  // 8192
    gemm_bf16_nt<float, 4><<<waves * 32 / 256, 256, 0, stream>>>(
        ob, Woutt, out, B * S, DIM, INNER);
  }
}
